// GatedS6Layer_15496242004271
// MI455X (gfx1250) — matrix operand order repaired and checked by reading, not compiled, not run on hardware
//
#include <hip/hip_runtime.h>

// ---------------------------------------------------------------------------
// Problem dims (compile-time constants from the reference)
// ---------------------------------------------------------------------------
#define BATCH 8
#define SEQ   1024
#define HID   1024
#define STATE 128
#define DTRD  32
#define EPSLN 1e-5f

typedef __attribute__((ext_vector_type(16))) __bf16 v16bf;
typedef __attribute__((ext_vector_type(8)))  float  v8f;

// Packed f32x2 -> bf16x2: round-half-up then gather high halves with a single
// v_perm_b32 (selector picks bytes {hi[3],hi[2],lo[3],lo[2]}). 3 VALU ops.
__device__ __forceinline__ unsigned pack_bf16(float lo, float hi) {
    unsigned ul = __float_as_uint(lo) + 0x8000u;
    unsigned uh = __float_as_uint(hi) + 0x8000u;
    return __builtin_amdgcn_perm(uh, ul, 0x07060302u);
}

union AFrag { v16bf v; uint4 q[2]; };

// ---------------------------------------------------------------------------
// Tiled bf16 WMMA GEMM: Y[M,N] = X[M,K] @ W[N,K]^T + bias[N]
// Block = 256 threads (8 waves), 128x128 output tile, K-chunk 32.
// Wave grid 2(M)x4(N); each wave owns 4x2 16x16 accumulators.
// Software-pipelined: global loads for chunk k+1 issue before the WMMAs of
// chunk k (s_wait_loadcnt lands after a full WMMA phase + barrier of slack).
// Requires M%128==0, N%128==0, K%32==0 (true for all call sites).
// ---------------------------------------------------------------------------
__global__ __launch_bounds__(256)
void wmma_gemm_kernel(const float* __restrict__ X, const float* __restrict__ W,
                      const float* __restrict__ bias, float* __restrict__ Y,
                      int M, int N, int K)
{
    __shared__ __align__(16) unsigned int XsU[128 * 16];  // 128 rows x 32 bf16
    __shared__ __align__(16) unsigned int WsU[128 * 16];

    const int t     = threadIdx.x;
    const int lane  = t & 31;
    const int w     = t >> 5;       // 0..7
    const int waveM = w >> 2;       // 0..1 -> 64-row slab
    const int waveN = w & 3;        // 0..3 -> 32-col slab
    const int half  = lane >> 4;    // 0/1
    const int l16   = lane & 15;

    const int nBase = blockIdx.x * 128;
    const int mBase = blockIdx.y * 128;

    v8f acc[4][2] = {};

    const int rload = t >> 3;          // 0..31
    const int cload = (t & 7) * 4;     // 0..28 step 4

    const int nk = K / 32;
    float4 xpre[4], wpre[4];

    // prefetch chunk 0
    #pragma unroll
    for (int p = 0; p < 4; ++p) {
        int r = p * 32 + rload;
        xpre[p] = *(const float4*)(X + (size_t)(mBase + r) * K + cload);
        wpre[p] = *(const float4*)(W + (size_t)(nBase + r) * K + cload);
    }

    for (int kc = 0; kc < nk; ++kc) {
        // ---- convert prefetched f32 -> bf16 (v_perm_b32 pack), stage LDS ----
        #pragma unroll
        for (int p = 0; p < 4; ++p) {
            int r = p * 32 + rload;
            XsU[r * 16 + (cload >> 1)]     = pack_bf16(xpre[p].x, xpre[p].y);
            XsU[r * 16 + (cload >> 1) + 1] = pack_bf16(xpre[p].z, xpre[p].w);
            WsU[r * 16 + (cload >> 1)]     = pack_bf16(wpre[p].x, wpre[p].y);
            WsU[r * 16 + (cload >> 1) + 1] = pack_bf16(wpre[p].z, wpre[p].w);
        }
        __syncthreads();

        // ---- issue global loads for next chunk (overlap with WMMAs) ----
        if (kc + 1 < nk) {
            const int k0 = (kc + 1) * 32;
            #pragma unroll
            for (int p = 0; p < 4; ++p) {
                int r = p * 32 + rload;
                xpre[p] = *(const float4*)(X + (size_t)(mBase + r) * K + k0 + cload);
                wpre[p] = *(const float4*)(W + (size_t)(nBase + r) * K + k0 + cload);
            }
        }

        // ---- load fragments per CDNA5 16-bit WMMA layouts ----
        AFrag a[4], b[2];
        #pragma unroll
        for (int tm = 0; tm < 4; ++tm) {
            int row = waveM * 64 + tm * 16 + l16;            // A: M = lane%16
            const uint4* p = (const uint4*)&XsU[row * 16];
            a[tm].q[0] = p[half];          // K = 8*half + (0..7)
            a[tm].q[1] = p[2 + half];      // K = 16 + 8*half + (0..7)
        }
        #pragma unroll
        for (int tn = 0; tn < 2; ++tn) {
            int nrow = waveN * 32 + tn * 16 + l16;           // B: N = lane%16
            const uint4* p = (const uint4*)&WsU[nrow * 16];
            b[tn].q[0] = p[half];          // K = 8*half + (0..7)
            b[tn].q[1] = p[2 + half];      // K = 16 + 8*half + (0..7)
        }

        #pragma unroll
        for (int tm = 0; tm < 4; ++tm)
            #pragma unroll
            for (int tn = 0; tn < 2; ++tn)
                acc[tm][tn] = __builtin_amdgcn_wmma_f32_16x16x32_bf16(
                    false, a[tm].v, false, b[tn].v,
                    (short)0, acc[tm][tn], false, false);
        __syncthreads();
    }

    // ---- epilogue: bias + store (D layout: M = r + 8*half, N = lane%16) ----
    #pragma unroll
    for (int tn = 0; tn < 2; ++tn) {
        int gn = nBase + waveN * 32 + tn * 16 + l16;
        float bb = bias[gn];
        #pragma unroll
        for (int tm = 0; tm < 4; ++tm) {
            #pragma unroll
            for (int r = 0; r < 8; ++r) {
                int gm = mBase + waveM * 64 + tm * 16 + r + 8 * half;
                Y[(size_t)gm * N + gn] = acc[tm][tn][r] + bb;
            }
        }
    }
}

// ---------------------------------------------------------------------------
// Row-wise LayerNorm -> sigmoid -> gate (* mulsrc) [+ resid] ; one block/row
// ---------------------------------------------------------------------------
__global__ __launch_bounds__(256)
void ln_gate_kernel(const float* __restrict__ pre, const float* __restrict__ gamma,
                    const float* __restrict__ beta, const float* __restrict__ mulsrc,
                    const float* __restrict__ resid, float* __restrict__ out)
{
    __shared__ float redS[256];
    __shared__ float redQ[256];
    const int row = blockIdx.x;
    const int t = threadIdx.x;
    const size_t base = (size_t)row * HID;

    float v[4];
    float s = 0.f, q = 0.f;
    #pragma unroll
    for (int e = 0; e < 4; ++e) {
        v[e] = pre[base + t + e * 256];
        s += v[e];
        q += v[e] * v[e];
    }
    redS[t] = s; redQ[t] = q;
    __syncthreads();
    for (int off = 128; off > 0; off >>= 1) {
        if (t < off) { redS[t] += redS[t + off]; redQ[t] += redQ[t + off]; }
        __syncthreads();
    }
    float mu   = redS[0] * (1.0f / HID);
    float var  = redQ[0] * (1.0f / HID) - mu * mu;
    float rstd = rsqrtf(var + EPSLN);

    #pragma unroll
    for (int e = 0; e < 4; ++e) {
        int col = t + e * 256;
        float z = (v[e] - mu) * rstd * gamma[col] + beta[col];
        float g = 1.0f / (1.0f + __expf(-z));
        float o = mulsrc[base + col] * g;
        if (resid) o += resid[base + col];
        out[base + col] = o;
    }
}

// ---------------------------------------------------------------------------
// dtr = x_gated @ dtr_W^T + dtr_b      (8192 x 32, K=1024) — VALU, float4
// ---------------------------------------------------------------------------
__global__ __launch_bounds__(256)
void dtr_kernel(const float* __restrict__ xg, const float* __restrict__ Wd,
                const float* __restrict__ bd, float* __restrict__ out)
{
    int idx = blockIdx.x * 256 + threadIdx.x;
    int m = idx >> 5, n = idx & 31;
    const float4* xr = (const float4*)(xg + (size_t)m * HID);
    const float4* wr = (const float4*)(Wd + (size_t)n * HID);
    float s = bd[n];
    #pragma unroll 4
    for (int k = 0; k < HID / 4; ++k) {
        float4 a = xr[k], b = wr[k];
        s += a.x * b.x + a.y * b.y + a.z * b.z + a.w * b.w;
    }
    out[idx] = s;
}

// ---------------------------------------------------------------------------
// dt = softplus(dtr @ dtp_W^T + dtp_b) (8192 x 128, K=32) — VALU
// ---------------------------------------------------------------------------
__global__ __launch_bounds__(256)
void dt_kernel(const float* __restrict__ dtr, const float* __restrict__ Wp,
               const float* __restrict__ bp, float* __restrict__ out)
{
    int idx = blockIdx.x * 256 + threadIdx.x;
    int m = idx >> 7, n = idx & 127;
    const float* xr = dtr + (size_t)m * DTRD;
    const float* wr = Wp + (size_t)n * DTRD;
    float s = bp[n];
    #pragma unroll
    for (int k = 0; k < DTRD; ++k) s += xr[k] * wr[k];
    out[idx] = (s > 20.f) ? s : log1pf(__expf(s));
}

// ---------------------------------------------------------------------------
// Sequential SSM scan: state_i <- sum_j exp(dt_i * A_ij) * state_j + B_i
// One block per batch; 512 threads = 128 state rows x 4 j-partitions.
// A-row slice lives in VGPRs (full unroll); state broadcast via LDS.
// ---------------------------------------------------------------------------
#define PARTS 4
__global__ __launch_bounds__(512)
void scan_kernel(const float* __restrict__ A, const float* __restrict__ dt,
                 const float* __restrict__ Bt, float* __restrict__ states,
                 float* __restrict__ finalst)
{
    __shared__ float st[STATE];
    __shared__ float partial[PARTS * STATE];
    const int b    = blockIdx.x;
    const int t    = threadIdx.x;
    const int i    = t & (STATE - 1);
    const int part = t >> 7;                  // 0..3
    const int j0   = part * (STATE / PARTS);  // 32-wide slice

    float Arow[STATE / PARTS];
    #pragma unroll
    for (int j = 0; j < STATE / PARTS; ++j) Arow[j] = A[i * STATE + j0 + j];

    if (t < STATE) st[i] = 0.f;
    __syncthreads();

    for (int ts = 0; ts < SEQ; ++ts) {
        size_t o = ((size_t)b * SEQ + ts) * STATE;
        float d = dt[o + i];
        float p = 0.f;
        #pragma unroll
        for (int j = 0; j < STATE / PARTS; ++j)
            p += __expf(d * Arow[j]) * st[j0 + j];
        partial[part * STATE + i] = p;
        __syncthreads();
        if (part == 0) {
            float s = Bt[o + i] + partial[i] + partial[STATE + i]
                     + partial[2 * STATE + i] + partial[3 * STATE + i];
            st[i] = s;
            states[o + i] = s;
        }
        __syncthreads();
    }
    if (t < STATE) finalst[b * STATE + i] = st[i];
}

// ---------------------------------------------------------------------------
extern "C" void kernel_launch(void* const* d_in, const int* in_sizes, int n_in,
                              void* d_out, int out_size, void* d_ws, size_t ws_size,
                              hipStream_t stream)
{
    (void)in_sizes; (void)n_in; (void)out_size; (void)ws_size;

    const float* x     = (const float*)d_in[0];
    const float* A     = (const float*)d_in[1];
    const float* B_W   = (const float*)d_in[2];
    const float* B_b   = (const float*)d_in[3];
    const float* C_W   = (const float*)d_in[4];
    const float* C_b   = (const float*)d_in[5];
    const float* dtp_W = (const float*)d_in[6];
    const float* dtp_b = (const float*)d_in[7];
    const float* dtr_W = (const float*)d_in[8];
    const float* dtr_b = (const float*)d_in[9];
    const float* ig_W  = (const float*)d_in[10];
    const float* ig_b  = (const float*)d_in[11];
    const float* ig_g  = (const float*)d_in[12];
    const float* ig_be = (const float*)d_in[13];
    const float* og_W  = (const float*)d_in[14];
    const float* og_b  = (const float*)d_in[15];
    const float* og_g  = (const float*)d_in[16];
    const float* og_be = (const float*)d_in[17];
    const float* res_W = (const float*)d_in[18];
    const float* res_b = (const float*)d_in[19];

    float* out    = (float*)d_out;                       // (8,1024,1024)
    float* fstate = out + (size_t)BATCH * SEQ * HID;     // (8,128) tail

    const int Mrows = BATCH * SEQ;                       // 8192
    float* buf0 = (float*)d_ws;                          // 8M f32 (ig->x_gated->og)
    float* buf1 = buf0 + (size_t)Mrows * HID;            // 8M f32 (ssm_out)
    float* dtrb = buf1 + (size_t)Mrows * HID;            // 8192*32
    float* dtb  = dtrb + (size_t)Mrows * DTRD;           // 8192*128
    float* Btb  = dtb  + (size_t)Mrows * STATE;          // 8192*128
    float* stb  = Btb  + (size_t)Mrows * STATE;          // 8192*128

    dim3 blk(256);

    // 1) ig pre-LN = x @ ig_W^T + ig_b
    wmma_gemm_kernel<<<dim3(HID / 128, Mrows / 128), blk, 0, stream>>>(
        x, ig_W, ig_b, buf0, Mrows, HID, HID);
    // 2) x_gated = x * sigmoid(LN(buf0))  (in place)
    ln_gate_kernel<<<dim3(Mrows), blk, 0, stream>>>(buf0, ig_g, ig_be, x, nullptr, buf0);
    // 3) dtr
    dtr_kernel<<<dim3(Mrows * DTRD / 256), blk, 0, stream>>>(buf0, dtr_W, dtr_b, dtrb);
    // 4) dt = softplus(...)
    dt_kernel<<<dim3(Mrows * STATE / 256), blk, 0, stream>>>(dtrb, dtp_W, dtp_b, dtb);
    // 5) B_t = x_gated @ B_W^T + B_b
    wmma_gemm_kernel<<<dim3(STATE / 128, Mrows / 128), blk, 0, stream>>>(
        buf0, B_W, B_b, Btb, Mrows, STATE, HID);
    // 6) sequential recurrence -> states + final_state
    scan_kernel<<<dim3(BATCH), dim3(512), 0, stream>>>(A, dtb, Btb, stb, fstate);
    // 7) ssm_out = states @ C_W^T + C_b
    wmma_gemm_kernel<<<dim3(HID / 128, Mrows / 128), blk, 0, stream>>>(
        stb, C_W, C_b, buf1, Mrows, HID, STATE);
    // 8) og pre-LN = ssm_out @ og_W^T + og_b   (overwrites buf0)
    wmma_gemm_kernel<<<dim3(HID / 128, Mrows / 128), blk, 0, stream>>>(
        buf1, og_W, og_b, buf0, Mrows, HID, HID);
    // 9) residual = x @ res_W^T + res_b  -> directly into d_out
    wmma_gemm_kernel<<<dim3(HID / 128, Mrows / 128), blk, 0, stream>>>(
        x, res_W, res_b, out, Mrows, HID, HID);
    // 10) out = ssm_out * sigmoid(LN(og)) + residual(in out)
    ln_gate_kernel<<<dim3(Mrows), blk, 0, stream>>>(buf0, og_g, og_be, buf1, out, out);
}
